// SingleOutModel_1786706395636
// MI455X (gfx1250) — matrix operand order repaired and checked by reading, not compiled, not run on hardware
//
#include <hip/hip_runtime.h>

typedef __attribute__((ext_vector_type(16))) _Float16 v16h;
typedef __attribute__((ext_vector_type(8)))  float    v8f;

#define NQ      6
#define DIM     64
#define NLAYERS 2
#define NCLS    7
#define WAVES_PER_BLOCK 8

// ---------------------------------------------------------------------------
// Kernel 1: build A = Re(U^H Z0 U) (64x64 fp32) from the 12 circuit angles,
// then pack A into WMMA B-operand f16 fragments (hi + lo error split).
// B-operand k order used here (contiguous, same as A) for V_WMMA_F32_16X16X32_F16 (32x16 per fragment):
//   lane l: N = l%16, halves h=0..15 -> K = ksBase + (l>=16 ? 16 : 0) + h
// Fragment order in ws: frag f = ks*4 + nt, element ((f*32 + l)*16 + h).
// ---------------------------------------------------------------------------
__global__ __launch_bounds__(64) void build_quadform(
    const float* __restrict__ qw,
    _Float16* __restrict__ bh, _Float16* __restrict__ bl)
{
  __shared__ float Ur[DIM][DIM];   // [column][k]  (column of U per thread)
  __shared__ float Ui[DIM][DIM];
  __shared__ float Amat[DIM][DIM]; // A[k][n]
  const int t = threadIdx.x;

  for (int k = 0; k < DIM; ++k) { Ur[t][k] = (k == t) ? 1.f : 0.f; Ui[t][k] = 0.f; }

  // qubit w <-> bit (5-w) of the flat index (qubit 0 is the MSB / measured one)
  for (int l = 0; l < NLAYERS; ++l) {
    for (int w = 0; w < NQ; ++w) {           // RX(theta) on qubit w
      float th = qw[l * NQ + w] * 0.5f;
      float c = __cosf(th), s = __sinf(th);
      int st = 1 << (5 - w);
      for (int k = 0; k < DIM; ++k) {
        if (k & st) continue;
        float ar = Ur[t][k],      ai = Ui[t][k];
        float br = Ur[t][k + st], bi = Ui[t][k + st];
        // U = [[c, -i s], [-i s, c]]
        Ur[t][k]      =  c * ar + s * bi;
        Ui[t][k]      =  c * ai - s * br;
        Ur[t][k + st] =  s * ai + c * br;
        Ui[t][k + st] = -s * ar + c * bi;
      }
    }
    for (int w = 0; w < NQ; ++w) {           // CNOT(w -> (w+1)%6)
      int cb = 1 << (5 - w);
      int tb = 1 << (5 - ((w + 1) % NQ));
      for (int k = 0; k < DIM; ++k) {
        if ((k & cb) && !(k & tb)) {
          int k2 = k | tb;
          float x;
          x = Ur[t][k]; Ur[t][k] = Ur[t][k2]; Ur[t][k2] = x;
          x = Ui[t][k]; Ui[t][k] = Ui[t][k2]; Ui[t][k2] = x;
        }
      }
    }
  }
  __syncthreads();

  // A[i][j] = sum_k z_k * ( Re U[k][i] Re U[k][j] + Im U[k][i] Im U[k][j] )
  for (int j = 0; j < DIM; ++j) {
    float acc = 0.f;
    for (int k = 0; k < DIM; ++k) {
      float zk = (k & 32) ? -1.f : 1.f;
      acc += zk * (Ur[t][k] * Ur[j][k] + Ui[t][k] * Ui[j][k]);
    }
    Amat[t][j] = acc;
  }
  __syncthreads();

  // Pack 8 fragments (ks=0..1, nt=0..3) x 32 lanes x 16 halves.
  for (int idx = t; idx < 8 * 32 * 16; idx += 64) {
    int h  = idx & 15;
    int l  = (idx >> 4) & 31;
    int f  = idx >> 9;
    int ks = f >> 2, nt = f & 3;
    int K  = ks * 32 + ((l >= 16) ? 16 : 0) + h;
    int N  = nt * 16 + (l & 15);
    float v = Amat[K][N];
    _Float16 vh = (_Float16)v;
    bh[idx] = vh;
    bl[idx] = (_Float16)(v - (float)vh);
  }
}

// ---------------------------------------------------------------------------
// Kernel 2: per wave, 16 samples. Build X tile directly in the WMMA A-operand
// layout (f16 hi/lo split) + fp32 copy in LDS; Y = X*A via 24 v_wmma per tile
// (3-term compensated f16 product); z = <x,y> in fp32; out = z*hw + hb.
// ---------------------------------------------------------------------------
__global__ __launch_bounds__(256) void qforward(
    const float* __restrict__ xc,   // (B,8)
    const float* __restrict__ xa,   // (B,4), only cols 0..2 used
    const _Float16* __restrict__ bh,
    const _Float16* __restrict__ bl,
    const float* __restrict__ hw,   // (7)
    const float* __restrict__ hb,   // (7)
    float* __restrict__ out,        // (B,7)
    int n_tiles)
{
  __shared__ float xs[WAVES_PER_BLOCK][16][65];   // fp32 X tile, padded

  const int lane = threadIdx.x & 31;
  const int wave = threadIdx.x >> 5;
  const int tile = blockIdx.x * WAVES_PER_BLOCK + wave;
  if (tile >= n_tiles) return;

  const int  m  = lane & 15;        // row of the 16-sample tile
  const bool hi = lane >= 16;
  const int  s  = tile * 16 + m;    // sample index

  // --- per-sample scalars (duplicated across the lane pair; cheap VALU) ---
  const float4* xc4 = (const float4*)(xc + (size_t)s * 8);
  float4 p0 = xc4[0], p1 = xc4[1];
  float nrm = p0.x*p0.x + p0.y*p0.y + p0.z*p0.z + p0.w*p0.w
            + p1.x*p1.x + p1.y*p1.y + p1.z*p1.z + p1.w*p1.w;
  float inv = 1.0f / sqrtf(nrm);
  float amp[8] = { p0.x*inv, p0.y*inv, p0.z*inv, p0.w*inv,
                   p1.x*inv, p1.y*inv, p1.z*inv, p1.w*inv };
  float h0 = xa[(size_t)s * 4 + 0] * 0.5f;
  float h1 = xa[(size_t)s * 4 + 1] * 0.5f;
  float h2 = xa[(size_t)s * 4 + 2] * 0.5f;
  float t3[2] = { __cosf(h0), __sinf(h0) };
  float t4[2] = { __cosf(h1), __sinf(h1) };
  float t5[2] = { __cosf(h2), __sinf(h2) };

  // --- build X as WMMA A operand (16-bit A 16x32), contiguous k order as in B:
  //     K(h) = ks*32 + h + (lane>=16 ? 16 : 0)
  v16h ah[2], al[2];
#pragma unroll
  for (int ks = 0; ks < 2; ++ks) {
#pragma unroll
    for (int h = 0; h < 16; ++h) {
      int k = ks * 32 + h + (hi ? 16 : 0);
      float v = amp[k >> 3] * t3[(k >> 2) & 1] * t4[(k >> 1) & 1] * t5[k & 1];
      xs[wave][m][k] = v;                       // exact fp32 copy for the dot
      _Float16 vh = (_Float16)v;
      ah[ks][h] = vh;
      al[ks][h] = (_Float16)(v - (float)vh);
    }
  }

  // --- GEMM tile: Y(16x64) = X(16x64) * A(64x64), compensated f16 WMMA ---
  const v16h* BH = (const v16h*)bh;
  const v16h* BL = (const v16h*)bl;
  v8f acc[4];
#pragma unroll
  for (int nt = 0; nt < 4; ++nt) {
    v8f c = {};
#pragma unroll
    for (int ks = 0; ks < 2; ++ks) {
      v16h Bh = BH[(ks * 4 + nt) * 32 + lane];
      v16h Bl = BL[(ks * 4 + nt) * 32 + lane];
      c = __builtin_amdgcn_wmma_f32_16x16x32_f16(false, al[ks], false, Bh,
                                                 (short)0, c, false, false);
      c = __builtin_amdgcn_wmma_f32_16x16x32_f16(false, ah[ks], false, Bl,
                                                 (short)0, c, false, false);
      c = __builtin_amdgcn_wmma_f32_16x16x32_f16(false, ah[ks], false, Bh,
                                                 (short)0, c, false, false);
    }
    acc[nt] = c;
  }

  // --- z = <x, y> per row. D layout: VGPR r <-> row r (+8 for lanes>=16),
  //     N = nt*16 + lane%16. Partial products vs fp32 x from LDS.
  float p[8];
#pragma unroll
  for (int r = 0; r < 8; ++r) p[r] = 0.f;
#pragma unroll
  for (int nt = 0; nt < 4; ++nt) {
    int col = nt * 16 + m;
#pragma unroll
    for (int r = 0; r < 8; ++r) {
      int M = r + (hi ? 8 : 0);
      p[r] += acc[nt][r] * xs[wave][M][col];
    }
  }
  // reduce each p[r] across the 16-lane group (xor masks stay in-group)
#pragma unroll
  for (int mask = 1; mask < 16; mask <<= 1) {
#pragma unroll
    for (int r = 0; r < 8; ++r) p[r] += __shfl_xor(p[r], mask, 32);
  }

  // --- write: lanes 0..7 -> rows 0..7, lanes 16..23 -> rows 8..15 ---
  int rsel = lane & 15;
  if (rsel < 8) {
    float z = p[0];
#pragma unroll
    for (int r = 1; r < 8; ++r) z = (rsel == r) ? p[r] : z;
    int row = rsel + (hi ? 8 : 0);
    size_t so = (size_t)(tile * 16 + row) * NCLS;
#pragma unroll
    for (int c = 0; c < NCLS; ++c) out[so + c] = z * hw[c] + hb[c];
  }
}

// ---------------------------------------------------------------------------
extern "C" void kernel_launch(void* const* d_in, const int* in_sizes, int n_in,
                              void* d_out, int out_size, void* d_ws, size_t ws_size,
                              hipStream_t stream) {
  const float* x_cont = (const float*)d_in[0];
  const float* x_cat  = (const float*)d_in[1];
  const float* qw     = (const float*)d_in[2];
  const float* hw     = (const float*)d_in[3];
  const float* hb     = (const float*)d_in[4];
  float* out = (float*)d_out;

  _Float16* bh = (_Float16*)d_ws;            // 8 frags * 32 lanes * 16 halves
  _Float16* bl = bh + 8 * 32 * 16;           // lo-split fragments (16 KB total)

  int B = in_sizes[0] / 8;
  int n_tiles = B / 16;                      // B = 262144 -> 16384 tiles
  int blocks  = (n_tiles + WAVES_PER_BLOCK - 1) / WAVES_PER_BLOCK;

  build_quadform<<<1, 64, 0, stream>>>(qw, bh, bl);
  qforward<<<blocks, 256, 0, stream>>>(x_cont, x_cat, bh, bl, hw, hb, out, n_tiles);
}
